// ShowTellNet_63410897158252
// MI455X (gfx1250) — matrix operand order repaired and checked by reading, not compiled, not run on hardware
//
#include <hip/hip_runtime.h>
#include <hip/hip_bf16.h>
#include <math.h>

// ---------------------------------------------------------------------------
// ShowAndTell forward on gfx1250 (MI455X), wave32 + WMMA f16 (f32 accum).
//   CNN:   3x [GEMM(A @ W^T)+bias] -> BN(train) -> ReLU      (float4 loads)
//   xU:    (emb gather fused) xe @ U  -> [48,32,2048]
//   LSTM:  single 32-wave workgroup; h in LDS (f16, padded), c in registers,
//          gate tiles co-located per wave; V pre-packed to f16 fragments
//   OUT:   TDM (tensor_load_to_lds) stages the A tile into LDS with HW row
//          padding; Wout pre-packed to f16 fragments; then row softmax.
// ---------------------------------------------------------------------------

typedef __attribute__((ext_vector_type(16))) _Float16     v16h;
typedef __attribute__((ext_vector_type(8)))  _Float16     v8h;
typedef __attribute__((ext_vector_type(8)))  float        v8f;
typedef __attribute__((ext_vector_type(4)))  float        v4f;
typedef __attribute__((ext_vector_type(4)))  unsigned int v4u;
typedef __attribute__((ext_vector_type(4)))  int          v4i;
typedef __attribute__((ext_vector_type(8)))  int          v8i;

#define BS      32
#define SEQ     48
#define VOCAB   32529
#define HH      512
#define IMG_F   4096
#define G4H     2048
#define NTILE_V ((VOCAB + 15) / 16)      // 2034
#define HPAD    520                      // padded LDS row stride in f16 elems

// A-matrix (16x32 f16) slot -> k offset within the 32-wide K slab.
__device__ __forceinline__ int a_slot_k(int half, int i) {
    return (i < 8) ? (half * 8 + i) : (16 + half * 8 + (i - 8));
}
// B-matrix (32x16 f16) slot -> k offset.
__device__ __forceinline__ int b_slot_k(int half, int i) {
    return half * 16 + i;
}

__device__ __forceinline__ v8f vzero8() { v8f z = {}; return z; }

__device__ __forceinline__ v16h cat8(v8h lo, v8h hi) {
    v16h r;
#pragma unroll
    for (int i = 0; i < 8; ++i) { r[i] = lo[i]; r[i + 8] = hi[i]; }
    return r;
}

__device__ __forceinline__ float sigmoidf_(float x) {
    return 1.0f / (1.0f + __expf(-x));
}

// ---------------------------------------------------------------------------
// TDM: stage a 16 x 512 f16 tile (rows from `gsrc`) into LDS at `ldsAddr`,
// with HW padding of 4 DWORDs (16 B) after every 256 DWORDs (1024 B row) so
// rows land at stride HPAD=520 halfs. D# layout per CDNA5 ISA.
// ---------------------------------------------------------------------------
__device__ __forceinline__ void tdm_load_tile16x512_f16(const _Float16* gsrc,
                                                        unsigned int ldsAddr) {
    unsigned long long ga = (unsigned long long)(const void*)gsrc;
    v4u g0 = { 1u,                                   // count=1 (valid user D#)
               ldsAddr,                              // lds_addr [63:32]
               (unsigned int)(ga & 0xffffffffu),     // global_addr lo
               (unsigned int)((ga >> 32) & 0x01ffffffu) | (2u << 30) }; // +type=2
    v8i g1 = { (int)((1u << 16)    // data_size = 1 -> 2 bytes
                  | (1u << 20)     // pad_enable
                  | (7u << 22)     // pad_interval: 256 DWORDs (1024 B)
                  | (3u << 25)),   // pad_amount: 4 DWORDs (16 B)
               (int)(512u << 16),  // tensor_dim0 = 512 elems (bits 79:48)
               (int)(16u << 16),   // tensor_dim1 = 16        (bits 111:80)
               (int)(512u << 16),  // tile_dim0   = 512       (bits 127:112)
               16,                 // tile_dim1=16, tile_dim2=0
               512,                // tensor_dim0_stride = 512 elems
               0, 0 };             // tensor_dim1_stride = 0 (2D)
    v4i gz = { 0, 0, 0, 0 };
#if defined(__clang_major__) && __clang_major__ >= 23
    v8i gz8 = { 0, 0, 0, 0, 0, 0, 0, 0 };
    __builtin_amdgcn_tensor_load_to_lds(g0, g1, gz, gz, gz8, 0);
#else
    __builtin_amdgcn_tensor_load_to_lds(g0, g1, gz, gz, 0);
#endif
    __builtin_amdgcn_s_wait_tensorcnt((short)0);
}

// ---------------------------------------------------------------------------
// Pack V (f32 [512,2048]) into f16 B-fragments:
//   Vpk[((ks*128 + nt)*32 + L)*16 + i] = V[(ks*32 + (L>>4)*16 + i)*2048 + nt*16 + (L&15)]
// so a wave's B fragment is two contiguous b128 loads per lane.
// ---------------------------------------------------------------------------
__global__ void pack_v_kernel(const float* __restrict__ V,
                              _Float16* __restrict__ Vpk) {
    int t  = blockIdx.x * blockDim.x + threadIdx.x;   // 16*128*32 = 65536
    int L  = t & 31;
    int nt = (t >> 5) & 127;
    int ks = t >> 12;
    const float* src = V + (size_t)(ks * 32 + (L >> 4) * 16) * G4H + nt * 16 + (L & 15);
    _Float16* dst = Vpk + (size_t)t * 16;
#pragma unroll
    for (int i = 0; i < 16; ++i) dst[i] = (_Float16)src[(size_t)i * G4H];
}

// Same packing for Wout (f32 [512, 32529]) -> Wpk f16, column-clamped.
__global__ void pack_wout_kernel(const float* __restrict__ Wout,
                                 _Float16* __restrict__ Wpk) {
    size_t t = (size_t)blockIdx.x * blockDim.x + threadIdx.x;  // 16*2034*32
    int    L  = (int)(t & 31);
    size_t nt = (t >> 5) % NTILE_V;
    size_t ks = t / ((size_t)NTILE_V * 32);
    int n = (int)nt * 16 + (L & 15);
    if (n >= VOCAB) n = VOCAB - 1;
    const float* src = Wout + (ks * 32 + (size_t)(L >> 4) * 16) * VOCAB + n;
    _Float16* dst = Wpk + t * 16;
#pragma unroll
    for (int i = 0; i < 16; ++i) dst[i] = (_Float16)src[(size_t)i * VOCAB];
}

// ---------------------------------------------------------------------------
// Generic C[M,N] = A[M,K](f32) @ W[N,K]^T (f32) + bias. One 16x16 tile/wave.
// A and W are contiguous along K -> float4 loads + packed f16 converts.
// grid = (ceil(Ntiles/8), M/16), block = 256 (8 waves).
// ---------------------------------------------------------------------------
__global__ __launch_bounds__(256)
void gemm_awt_bias_kernel(const float* __restrict__ A,
                          const float* __restrict__ W,
                          const float* __restrict__ bias,
                          float* __restrict__ C, int N, int K) {
    const int lane = threadIdx.x & 31;
    const int wave = threadIdx.x >> 5;
    const int nt   = blockIdx.x * 8 + wave;
    if (nt >= (N >> 4)) return;
    const int mBase = blockIdx.y * 16;

    const int mA   = lane & 15;
    const int half = lane >> 4;
    const int nB   = nt * 16 + (lane & 15);

    v8f acc = vzero8();
    const float* arow = A + (size_t)(mBase + mA) * K;
    const float* wrow = W + (size_t)nB * K;

    for (int k0 = 0; k0 < K; k0 += 32) {
        const v4f* ap0 = (const v4f*)(arow + k0 + half * 16);
        const v4f* ap1 = (const v4f*)(arow + k0 + half * 16 + 8);
        v4f a0 = ap0[0], a1 = ap0[1], a2 = ap1[0], a3 = ap1[1];
        const v4f* wp = (const v4f*)(wrow + k0 + half * 16);
        v4f w0 = wp[0], w1 = wp[1], w2 = wp[2], w3 = wp[3];
        v16h af, bf;
#pragma unroll
        for (int i = 0; i < 4; ++i) {
            af[i]      = (_Float16)a0[i];
            af[4 + i]  = (_Float16)a1[i];
            af[8 + i]  = (_Float16)a2[i];
            af[12 + i] = (_Float16)a3[i];
            bf[i]      = (_Float16)w0[i];
            bf[4 + i]  = (_Float16)w1[i];
            bf[8 + i]  = (_Float16)w2[i];
            bf[12 + i] = (_Float16)w3[i];
        }
        acc = __builtin_amdgcn_wmma_f32_16x16x32_f16(
            false, af, false, bf, (short)0, acc, false, false);
    }

    const float bo   = bias[nB];
    const int   mAdd = (lane >= 16) ? 8 : 0;
#pragma unroll
    for (int r = 0; r < 8; ++r)
        C[(size_t)(mBase + r + mAdd) * N + nB] = acc[r] + bo;
}

// ---------------------------------------------------------------------------
// Train-mode BatchNorm over batch dim (32 rows) + ReLU, in place.
// ---------------------------------------------------------------------------
__global__ void bn_relu_kernel(float* __restrict__ h,
                               const float* __restrict__ g,
                               const float* __restrict__ be, int N) {
    int n = blockIdx.x * blockDim.x + threadIdx.x;
    if (n >= N) return;
    float sum = 0.f, sq = 0.f;
#pragma unroll 4
    for (int m = 0; m < BS; ++m) {
        float v = h[(size_t)m * N + n];
        sum += v; sq += v * v;
    }
    float mu  = sum * (1.0f / BS);
    float var = sq * (1.0f / BS) - mu * mu;
    float sc  = g[n] * rsqrtf(var + 1e-5f);
    float sh  = be[n];
#pragma unroll 4
    for (int m = 0; m < BS; ++m) {
        float v = (h[(size_t)m * N + n] - mu) * sc + sh;
        h[(size_t)m * N + n] = fmaxf(v, 0.f);
    }
}

// ---------------------------------------------------------------------------
// xU[s*32+b, n] = sum_k emb[x[b,s], k] * U[k, n]. Embedding gather fused into
// the A-fragment load (float4). grid = (128/8, 96), block = 256.
// ---------------------------------------------------------------------------
__global__ __launch_bounds__(256)
void embed_u_gemm_kernel(const int* __restrict__ x,
                         const float* __restrict__ emb,
                         const float* __restrict__ U,
                         float* __restrict__ xU) {
    const int lane = threadIdx.x & 31;
    const int wave = threadIdx.x >> 5;
    const int nt   = blockIdx.x * 8 + wave;       // 0..127
    const int mBase = blockIdx.y * 16;            // rows m = s*32 + b

    const int mA   = lane & 15;
    const int half = lane >> 4;
    const int m    = mBase + mA;
    const int s    = m >> 5;
    const int b    = m & 31;
    const int tok  = x[b * SEQ + s];
    const float* erow = emb + (size_t)tok * HH;
    const int nB = nt * 16 + (lane & 15);

    v8f acc = vzero8();
    for (int k0 = 0; k0 < HH; k0 += 32) {
        const v4f* ap0 = (const v4f*)(erow + k0 + half * 16);
        const v4f* ap1 = (const v4f*)(erow + k0 + half * 16 + 8);
        v4f a0 = ap0[0], a1 = ap0[1], a2 = ap1[0], a3 = ap1[1];
        v16h af, bf;
#pragma unroll
        for (int i = 0; i < 4; ++i) {
            af[i]      = (_Float16)a0[i];
            af[4 + i]  = (_Float16)a1[i];
            af[8 + i]  = (_Float16)a2[i];
            af[12 + i] = (_Float16)a3[i];
        }
#pragma unroll
        for (int i = 0; i < 16; ++i)
            bf[i] = (_Float16)U[(size_t)(k0 + b_slot_k(half, i)) * G4H + nB];
        acc = __builtin_amdgcn_wmma_f32_16x16x32_f16(
            false, af, false, bf, (short)0, acc, false, false);
    }
    const int mAdd = (lane >= 16) ? 8 : 0;
#pragma unroll
    for (int r = 0; r < 8; ++r)
        xU[(size_t)(mBase + r + mAdd) * G4H + nB] = acc[r];
}

// ---------------------------------------------------------------------------
// LSTM: one workgroup, 32 waves. Wave w owns j in [w*16, w*16+16): it computes
// gate tiles {w, w+32, w+64, w+96} of z = h@V + xU[s] + bg for both M-tiles,
// so i/f/g/o for a (b, j) land in the same lane; c never leaves registers.
// h lives in LDS as f16 at padded stride; V comes pre-packed (b128 B loads).
// ---------------------------------------------------------------------------
__global__ __launch_bounds__(1024)
void lstm_kernel(const float* __restrict__ feat,
                 const float* __restrict__ xU,
                 const float* __restrict__ bg,
                 const _Float16* __restrict__ Vpk,
                 _Float16* __restrict__ hs16,
                 float* __restrict__ hc_out) {
    __shared__ _Float16 hbuf[BS * HPAD];   // 32.5 KB

    const int lane = threadIdx.x & 31;
    const int w    = threadIdx.x >> 5;     // 0..31
    const int mAdd = (lane >= 16) ? 8 : 0;
    const int nIdx = lane & 15;
    const int j    = w * 16 + nIdx;        // hidden index owned by this lane

    const float bgi = bg[0 * HH + j];
    const float bgf = bg[1 * HH + j];
    const float bgg = bg[2 * HH + j];
    const float bgo = bg[3 * HH + j];

    float creg[2][8];
    float hlast[2][8];

    // ---- image step: h = gates(feat_rep + bg), c from zero ----
#pragma unroll
    for (int t = 0; t < 2; ++t)
#pragma unroll
        for (int r = 0; r < 8; ++r) {
            int b = t * 16 + r + mAdd;
            float fe = feat[b * HH + j];
            float ig = sigmoidf_(fe + bgi);
            float gg = tanhf(fe + bgg);
            float og = sigmoidf_(fe + bgo);
            float cc = ig * gg;                 // f*0 + i*g
            creg[t][r] = cc;
            float hh = og * tanhf(cc);
            hlast[t][r] = hh;
            hbuf[b * HPAD + j] = (_Float16)hh;
        }
    __syncthreads();

    const int mA   = lane & 15;
    const int half = lane >> 4;

    for (int s = 0; s < SEQ; ++s) {
        v8f acc[2][4];
#pragma unroll
        for (int t = 0; t < 2; ++t)
#pragma unroll
            for (int g = 0; g < 4; ++g) acc[t][g] = vzero8();

        for (int ks = 0; ks < 16; ++ks) {
            const int k0 = ks * 32;
            v16h a0 = cat8(*(const v8h*)(hbuf + (0  + mA) * HPAD + k0 + half * 16),
                           *(const v8h*)(hbuf + (0  + mA) * HPAD + k0 + half * 16 + 8));
            v16h a1 = cat8(*(const v8h*)(hbuf + (16 + mA) * HPAD + k0 + half * 16),
                           *(const v8h*)(hbuf + (16 + mA) * HPAD + k0 + half * 16 + 8));
#pragma unroll
            for (int g = 0; g < 4; ++g) {
                const v8h* bp = (const v8h*)(Vpk +
                    (((size_t)ks * 128 + (g * 32 + w)) * 32 + lane) * 16);
                v16h bf = cat8(bp[0], bp[1]);
                acc[0][g] = __builtin_amdgcn_wmma_f32_16x16x32_f16(
                    false, a0, false, bf, (short)0, acc[0][g], false, false);
                acc[1][g] = __builtin_amdgcn_wmma_f32_16x16x32_f16(
                    false, a1, false, bf, (short)0, acc[1][g], false, false);
            }
        }
        __syncthreads();   // every wave finished reading old h

#pragma unroll
        for (int t = 0; t < 2; ++t)
#pragma unroll
            for (int r = 0; r < 8; ++r) {
                int b = t * 16 + r + mAdd;
                const float* xrow = xU + ((size_t)s * BS + b) * G4H;
                float ig = sigmoidf_(acc[t][0][r] + xrow[0 * HH + j] + bgi);
                float fg = sigmoidf_(acc[t][1][r] + xrow[1 * HH + j] + bgf);
                float gg = tanhf    (acc[t][2][r] + xrow[2 * HH + j] + bgg);
                float og = sigmoidf_(acc[t][3][r] + xrow[3 * HH + j] + bgo);
                float cc = fg * creg[t][r] + ig * gg;
                creg[t][r] = cc;
                float hh = og * tanhf(cc);
                hlast[t][r] = hh;
                hbuf[b * HPAD + j] = (_Float16)hh;
                hs16[((size_t)b * SEQ + s) * HH + j] = (_Float16)hh;
            }
        __syncthreads();   // new h visible before next step's reads
    }

#pragma unroll
    for (int t = 0; t < 2; ++t)
#pragma unroll
        for (int r = 0; r < 8; ++r) {
            int b = t * 16 + r + mAdd;
            hc_out[(size_t)b * HH + j]           = hlast[t][r];
            hc_out[(size_t)BS * HH + b * HH + j] = creg[t][r];
        }
}

// ---------------------------------------------------------------------------
// logits[m, n] = hs16[m, :] @ Wout[:, n] + bout[n]   (m = b*48+s, fp32 out)
// A tile staged to LDS by the Tensor Data Mover (HW row padding -> stride 520).
// B comes from the pre-packed f16 Wpk (two b128 loads per lane per K slab).
// grid = (ceil(2034/8), 96), block = 256.
// ---------------------------------------------------------------------------
__global__ __launch_bounds__(256)
void out_gemm_packed_kernel(const _Float16* __restrict__ hs16,
                            const _Float16* __restrict__ Wpk,
                            const float* __restrict__ bout,
                            float* __restrict__ out) {
    __shared__ _Float16 tileA[16 * HPAD];   // 16.25 KB, TDM-padded rows

    const int lane = threadIdx.x & 31;
    const int wave = threadIdx.x >> 5;
    const int nt   = blockIdx.x * 8 + wave;
    const int ntc  = (nt < NTILE_V) ? nt : (NTILE_V - 1);
    const int mBase = blockIdx.y * 16;

    if (threadIdx.x < 32) {   // wave 0 drives the DMA
        tdm_load_tile16x512_f16(hs16 + (size_t)mBase * HH,
                                (unsigned int)(unsigned long long)(const void*)&tileA[0]);
    }
    __syncthreads();

    const int mA   = lane & 15;
    const int half = lane >> 4;
    const _Float16* arow = tileA + mA * HPAD;
    const _Float16* bbase = Wpk + (((size_t)ntc) * 32 + lane) * 16;
    const size_t slabStride = (size_t)NTILE_V * 32 * 16;   // f16 elems per K slab

    v8f acc = vzero8();
#pragma unroll 4
    for (int ks = 0; ks < 16; ++ks) {
        if (ks + 1 < 16)
            __builtin_prefetch(bbase + (size_t)(ks + 1) * slabStride, 0, 1);
        v16h af = cat8(*(const v8h*)(arow + ks * 32 + half * 16),
                       *(const v8h*)(arow + ks * 32 + half * 16 + 8));
        const v8h* bp = (const v8h*)(bbase + (size_t)ks * slabStride);
        v16h bf = cat8(bp[0], bp[1]);
        acc = __builtin_amdgcn_wmma_f32_16x16x32_f16(
            false, af, false, bf, (short)0, acc, false, false);
    }

    const int n = nt * 16 + (lane & 15);
    if (nt < NTILE_V && n < VOCAB) {
        const float bo   = bout[n];
        const int   mAdd = (lane >= 16) ? 8 : 0;
#pragma unroll
        for (int r = 0; r < 8; ++r)
            out[(size_t)(mBase + r + mAdd) * VOCAB + n] = acc[r] + bo;
    }
}

// Fallback when d_ws cannot hold the 33 MB packed Wout: inline f32->f16.
__global__ __launch_bounds__(256)
void out_gemm_inline_kernel(const _Float16* __restrict__ hs16,
                            const float* __restrict__ Wout,
                            const float* __restrict__ bout,
                            float* __restrict__ out) {
    __shared__ _Float16 tileA[16 * HPAD];
    const int lane = threadIdx.x & 31;
    const int wave = threadIdx.x >> 5;
    const int nt   = blockIdx.x * 8 + wave;
    const int mBase = blockIdx.y * 16;

    if (threadIdx.x < 32) {
        tdm_load_tile16x512_f16(hs16 + (size_t)mBase * HH,
                                (unsigned int)(unsigned long long)(const void*)&tileA[0]);
    }
    __syncthreads();

    const int mA   = lane & 15;
    const int half = lane >> 4;
    const int nB   = nt * 16 + (lane & 15);
    const int nBc  = (nB < VOCAB) ? nB : (VOCAB - 1);
    const _Float16* arow = tileA + mA * HPAD;

    v8f acc = vzero8();
    for (int k0 = 0; k0 < HH; k0 += 32) {
        v16h af = cat8(*(const v8h*)(arow + k0 + half * 16),
                       *(const v8h*)(arow + k0 + half * 16 + 8));
        v16h bf;
#pragma unroll
        for (int i = 0; i < 16; ++i)
            bf[i] = (_Float16)Wout[(size_t)(k0 + b_slot_k(half, i)) * VOCAB + nBc];
        acc = __builtin_amdgcn_wmma_f32_16x16x32_f16(
            false, af, false, bf, (short)0, acc, false, false);
    }

    const int n = nt * 16 + (lane & 15);
    if (nt < NTILE_V && n < VOCAB) {
        const float bo   = bout[n];
        const int   mAdd = (lane >= 16) ? 8 : 0;
#pragma unroll
        for (int r = 0; r < 8; ++r)
            out[(size_t)(mBase + r + mAdd) * VOCAB + n] = acc[r] + bo;
    }
}

// ---------------------------------------------------------------------------
// Row softmax over the vocab dim, in place. One 256-thread block per row.
// ---------------------------------------------------------------------------
__global__ __launch_bounds__(256)
void softmax_rows_kernel(float* __restrict__ out) {
    __shared__ float red[256];
    float* row = out + (size_t)blockIdx.x * VOCAB;
    const int tid = threadIdx.x;

    float mx = -1e30f;
    for (int i = tid; i < VOCAB; i += 256) mx = fmaxf(mx, row[i]);
    red[tid] = mx; __syncthreads();
    for (int off = 128; off; off >>= 1) {
        if (tid < off) red[tid] = fmaxf(red[tid], red[tid + off]);
        __syncthreads();
    }
    mx = red[0]; __syncthreads();

    float sum = 0.f;
    for (int i = tid; i < VOCAB; i += 256) sum += __expf(row[i] - mx);
    red[tid] = sum; __syncthreads();
    for (int off = 128; off; off >>= 1) {
        if (tid < off) red[tid] += red[tid + off];
        __syncthreads();
    }
    const float inv = 1.0f / red[0];
    __syncthreads();

    for (int i = tid; i < VOCAB; i += 256) row[i] = __expf(row[i] - mx) * inv;
}

// ---------------------------------------------------------------------------
extern "C" void kernel_launch(void* const* d_in, const int* in_sizes, int n_in,
                              void* d_out, int out_size, void* d_ws, size_t ws_size,
                              hipStream_t stream) {
    (void)in_sizes; (void)n_in; (void)out_size;

    const float* img  = (const float*)d_in[0];
    const int*   x    = (const int*)  d_in[1];
    const float* w1   = (const float*)d_in[2];
    const float* b1   = (const float*)d_in[3];
    const float* g1   = (const float*)d_in[4];
    const float* be1  = (const float*)d_in[5];
    const float* w2   = (const float*)d_in[6];
    const float* b2   = (const float*)d_in[7];
    const float* g2   = (const float*)d_in[8];
    const float* be2  = (const float*)d_in[9];
    const float* w3   = (const float*)d_in[10];
    const float* b3   = (const float*)d_in[11];
    const float* g3   = (const float*)d_in[12];
    const float* be3  = (const float*)d_in[13];
    const float* U    = (const float*)d_in[14];
    const float* V    = (const float*)d_in[15];
    const float* bg   = (const float*)d_in[16];
    const float* emb  = (const float*)d_in[17];
    const float* Wout = (const float*)d_in[18];
    const float* bout = (const float*)d_in[19];

    // workspace layout (bytes)
    char* ws = (char*)d_ws;
    float*    h1   = (float*)   (ws + 0);          // 32*2048 f32      (256 KB)
    float*    h2   = (float*)   (ws + 262144);     // 32*1024 f32      (128 KB)
    float*    feat = (float*)   (ws + 393216);     // 32*512  f32      ( 64 KB)
    _Float16* Vpk  = (_Float16*)(ws + 458752);     // packed V f16     (  2 MB)
    float*    xU   = (float*)   (ws + 2555904);    // 48*32*2048 f32   (12.6 MB)
    _Float16* hs16 = (_Float16*)(ws + 15138816);   // 32*48*512 f16    ( 1.5 MB)
    _Float16* Wpk  = (_Float16*)(ws + 16711680);   // packed Wout f16  (33.3 MB)
    const size_t needWpk = 16711680ull + (size_t)16 * NTILE_V * 32 * 16 * 2;
    const bool   useWpk  = ws_size >= needWpk;

    float* out   = (float*)d_out;
    float* hcOut = out + (size_t)BS * SEQ * VOCAB;

    // one-time operand packing (f32 -> fragment-native f16)
    pack_v_kernel<<<256, 256, 0, stream>>>(V, Vpk);
    if (useWpk) {
        const int tot = 16 * NTILE_V * 32;   // 1,041,408 threads
        pack_wout_kernel<<<(tot + 255) / 256, 256, 0, stream>>>(Wout, Wpk);
    }

    // CNN
    gemm_awt_bias_kernel<<<dim3(2048 / 16 / 8, 2), 256, 0, stream>>>(img, w1, b1, h1, 2048, IMG_F);
    bn_relu_kernel<<<(2048 + 255) / 256, 256, 0, stream>>>(h1, g1, be1, 2048);
    gemm_awt_bias_kernel<<<dim3(1024 / 16 / 8, 2), 256, 0, stream>>>(h1, w2, b2, h2, 1024, 2048);
    bn_relu_kernel<<<(1024 + 255) / 256, 256, 0, stream>>>(h2, g2, be2, 1024);
    gemm_awt_bias_kernel<<<dim3(512 / 16 / 8, 2), 256, 0, stream>>>(h2, w3, b3, feat, 512, 1024);
    bn_relu_kernel<<<(512 + 255) / 256, 256, 0, stream>>>(feat, g3, be3, 512);

    // xU = (emb[x]) @ U, stored [s][b][4H]
    embed_u_gemm_kernel<<<dim3(128 / 8, 96), 256, 0, stream>>>(x, emb, U, xU);

    // recurrent LSTM (single 32-wave workgroup), writes hs16 + final (h,c)
    lstm_kernel<<<1, 1024, 0, stream>>>(feat, xU, bg, Vpk, hs16, hcOut);

    // logits + softmax
    if (useWpk)
        out_gemm_packed_kernel<<<dim3((NTILE_V + 7) / 8, 96), 256, 0, stream>>>(hs16, Wpk, bout, out);
    else
        out_gemm_inline_kernel<<<dim3((NTILE_V + 7) / 8, 96), 256, 0, stream>>>(hs16, Wout, bout, out);
    softmax_rows_kernel<<<BS * SEQ, 256, 0, stream>>>(out);
}
